// MinGRUPolicyBackbone_57698590655153
// MI455X (gfx1250) — matrix operand order repaired and checked by reading, not compiled, not run on hardware
//
#include <hip/hip_runtime.h>
#include <hip/hip_bf16.h>
#include <math.h>

// ---------------------------------------------------------------------------
// MinGRU policy backbone, fused for MI455X (gfx1250, wave32, WMMA).
// B=16, S=4096, D_IN=64, D=512, DI=768.
// Only the LAST hidden state is needed -> segmented log-space reduction.
// hg = obs @ (W_emb @ W_hg) + b_emb @ W_hg   (linears folded, K=64)
// ---------------------------------------------------------------------------

typedef __attribute__((ext_vector_type(16))) _Float16 v16h;
typedef __attribute__((ext_vector_type(8)))  _Float16 v8h;
typedef __attribute__((ext_vector_type(8)))  float    v8f;

#define NB   16      // batch
#define NS   4096    // sequence
#define KIN  64      // input dim
#define DD   512     // model dim
#define DI   768     // inner dim
#define NSEG 16      // segments over S
#define SLEN (NS / NSEG)   // 256 timesteps per segment
#define NIT  (SLEN / 16)   // 16 wmma iterations per segment
#define CH   32      // DI-channels per block
#define LDSW 36      // padded LDS row stride: 8*36 % 64 == 32 -> conflict-free

// softplus(x) = log(1+e^x), branchless; __logf(1+t) error < 1e-7 abs (t<=1)
__device__ __forceinline__ float softplus_f(float x) {
    return fmaxf(x, 0.f) + __logf(1.f + __expf(-fabsf(x)));
}

// logaddexp(a,b) = max(a,b) + log(1 + exp(-|a-b|)); safe when one side = -inf
__device__ __forceinline__ float logaddexp_f(float a, float b) {
    const float m = fmaxf(a, b);
    const float d = fabsf(a - b);
    return m + __logf(1.f + __expf(-d));
}

__device__ __forceinline__ v8f splat8(float x) {
    v8f r;
#pragma unroll
    for (int i = 0; i < 8; ++i) r[i] = x;
    return r;
}

// A-matrix (16x32 f16, MxK) in contiguous k order (same as load_b):
// lane<16 : M=lane,    elems 0..15 = K kb..kb+15, kb=0
// lane>=16: M=lane-16, same with kb=16           (kHalf adds 0 or 32)
__device__ __forceinline__ v16h load_a(const _Float16* __restrict__ base, int lane, int kHalf) {
    int m  = lane & 15;
    int kb = kHalf + ((lane & 16) ? 16 : 0);
    const _Float16* p = base + m * KIN + kb;
    v8h lo = *(const v8h*)(p);        // K = kb .. kb+7
    v8h hi = *(const v8h*)(p + 8);    // K = kb+8 .. kb+15
    return __builtin_shufflevector(lo, hi, 0,1,2,3,4,5,6,7,8,9,10,11,12,13,14,15);
}

// B-matrix (32x16 f16, KxN): lane<16: N=lane, K = kHalf..kHalf+15 (elems 0..15)
//                            lane>=16: N=lane-16, K = kHalf+16..kHalf+31
// Wt is W_comb transposed: Wt[n*64 + k] == W_comb[k][n]; 16 consecutive K are
// contiguous (32B aligned) -> single v16h load.
__device__ __forceinline__ v16h load_b(const _Float16* __restrict__ Wt, int ncol, int lane, int kHalf) {
    int n  = ncol + (lane & 15);
    int kb = kHalf + ((lane & 16) ? 16 : 0);
    return *(const v16h*)(Wt + n * KIN + kb);
}

__device__ __forceinline__ v8f wmma_pair(v16h a_lo, v16h a_hi, v16h b_lo, v16h b_hi, v8f c) {
    c = __builtin_amdgcn_wmma_f32_16x16x32_f16(false, a_lo, false, b_lo, (short)0, c, false, false);
    c = __builtin_amdgcn_wmma_f32_16x16x32_f16(false, a_hi, false, b_hi, (short)0, c, false, false);
    return c;
}

// --------------------------- prep kernels ----------------------------------

__global__ void cvt_obs_kernel(const float* __restrict__ obs, _Float16* __restrict__ o16, int n) {
    int i = blockIdx.x * blockDim.x + threadIdx.x;
    if (i < n) o16[i] = (_Float16)obs[i];
}

// W_comb^T (f16) [2*DI][KIN] and b_comb (f32) [2*DI]
__global__ void prep_kernel(const float* __restrict__ W_emb,   // [KIN][DD]
                            const float* __restrict__ b_emb,   // [DD]
                            const float* __restrict__ W_hg,    // [DD][2*DI]
                            _Float16* __restrict__ Wt,         // [2*DI][KIN]
                            float* __restrict__ bcomb) {       // [2*DI]
    int n = blockIdx.x;    // 0 .. 2*DI-1
    int k = threadIdx.x;   // 0 .. KIN-1
    float s = 0.f;
    for (int j = 0; j < DD; ++j) s += W_emb[k * DD + j] * W_hg[j * (2 * DI) + n];
    Wt[n * KIN + k] = (_Float16)s;
    if (k == 0) {
        float bs = 0.f;
        for (int j = 0; j < DD; ++j) bs += b_emb[j] * W_hg[j * (2 * DI) + n];
        bcomb[n] = bs;
    }
}

// --------------------------- main fused kernel -----------------------------
// grid = (DI/CH=24, NB=16, NSEG=16), block = 32 (one wave).
// Computes hidden/gate tiles with WMMA, scans 16 timesteps per iteration,
// emits per-segment (LA, LV) minGRU composition in log space.
//
// NOTE: no __syncthreads() — the block is a single wave32. The DS pipeline
// executes in order per wave (cross-lane visibility guaranteed), and each
// lane's LDS stores/loads may-alias (lane c writes and reads column c), so
// the compiler preserves program order. Avoiding the barrier removes the
// s_wait_loadcnt 0 it implies, letting the prefetched A tile land during the
// transcendental scan instead of stalling before the LDS stores.
__global__ __launch_bounds__(32) void mingru_scan_kernel(
        const _Float16* __restrict__ obs16,   // [NB][NS][KIN]
        const _Float16* __restrict__ Wt,      // [2*DI][KIN]
        const float*    __restrict__ bcomb,   // [2*DI]
        float* __restrict__ segLA,            // [NB][NSEG][DI]
        float* __restrict__ segLV) {          // [NB][NSEG][DI]
    __shared__ float lds_h[16 * LDSW];
    __shared__ float lds_g[16 * LDSW];

    const int lane = threadIdx.x;
    const int cg   = blockIdx.x;   // channel group
    const int b    = blockIdx.y;   // batch
    const int seg  = blockIdx.z;   // segment
    const int c0   = cg * CH;      // hidden cols c0..c0+31; gate cols DI+c0..

    // Loop-invariant B operands (W_comb^T tiles), K=64 split into two x32 halves.
    const v16h b_h0_lo = load_b(Wt, c0,           lane, 0);
    const v16h b_h0_hi = load_b(Wt, c0,           lane, 32);
    const v16h b_h1_lo = load_b(Wt, c0 + 16,      lane, 0);
    const v16h b_h1_hi = load_b(Wt, c0 + 16,      lane, 32);
    const v16h b_g0_lo = load_b(Wt, DI + c0,      lane, 0);
    const v16h b_g0_hi = load_b(Wt, DI + c0,      lane, 32);
    const v16h b_g1_lo = load_b(Wt, DI + c0 + 16, lane, 0);
    const v16h b_g1_hi = load_b(Wt, DI + c0 + 16, lane, 32);

    // Bias folded into accumulator init (same value down each column t).
    const int cl = lane & 15;
    const float bias_h0 = bcomb[c0 + cl];
    const float bias_h1 = bcomb[c0 + 16 + cl];
    const float bias_g0 = bcomb[DI + c0 + cl];
    const float bias_g1 = bcomb[DI + c0 + 16 + cl];

    const _Float16* obs_base = obs16 + ((size_t)b * NS + (size_t)seg * SLEN) * KIN;

    float la = 0.f;
    float lv = -INFINITY;

    // Software pipeline: A-tile loads for iteration it+1 are issued before the
    // scan of iteration it, so the 16-step transcendental scan hides latency.
    v16h a_lo = load_a(obs_base, lane, 0);
    v16h a_hi = load_a(obs_base, lane, 32);

    for (int it = 0; it < NIT; ++it) {
        v8f acc_h0 = wmma_pair(a_lo, a_hi, b_h0_lo, b_h0_hi, splat8(bias_h0));
        v8f acc_h1 = wmma_pair(a_lo, a_hi, b_h1_lo, b_h1_hi, splat8(bias_h1));
        v8f acc_g0 = wmma_pair(a_lo, a_hi, b_g0_lo, b_g0_hi, splat8(bias_g0));
        v8f acc_g1 = wmma_pair(a_lo, a_hi, b_g1_lo, b_g1_hi, splat8(bias_g1));

        // Prefetch next A tile (clamped -> uniform, no divergence, no OOB).
        {
            const int nit = (it + 1 < NIT) ? (it + 1) : it;
            const _Float16* nb = obs_base + (size_t)nit * 16 * KIN;
            a_lo = load_a(nb, lane, 0);
            a_hi = load_a(nb, lane, 32);
        }

        // C/D layout: vgpr r -> (t = r + (lane>=16 ? 8:0), n = lane&15)
        const int tb = (lane & 16) ? 8 : 0;
#pragma unroll
        for (int r = 0; r < 8; ++r) {
            lds_h[(tb + r) * LDSW + cl]      = acc_h0[r];
            lds_h[(tb + r) * LDSW + cl + 16] = acc_h1[r];
            lds_g[(tb + r) * LDSW + cl]      = acc_g0[r];
            lds_g[(tb + r) * LDSW + cl + 16] = acc_g1[r];
        }

        // Sequential minGRU log-space scan: lane = channel, 16 timesteps.
        // Branchless, 6 transcendentals per element:
        //   sp = softplus(g); lc = -sp; lval = g - sp + log_g(h)
        //   log_g(h) = h>=0 ? log(h+0.5) : h - log(1+e^h)
#pragma unroll
        for (int t = 0; t < 16; ++t) {
            const float h = lds_h[t * LDSW + lane];
            const float g = lds_g[t * LDSW + lane];

            const float sp = softplus_f(g);                 // exp + log
            const float lc = -sp;                           // log(1-z)

            const float hm  = fminf(h, 0.f);
            const float u   = (h >= 0.f) ? (h + 0.5f) : (1.f + __expf(hm));
            const float lgu = __logf(u);                    // exp + log (shared)
            const float lg  = (h >= 0.f) ? lgu : (hm - lgu);

            const float lval = g - sp + lg;                 // log z + log g(h)
            lv = logaddexp_f(lv + lc, lval);                // exp + log
            la += lc;
        }
    }

    const int oi = ((b * NSEG + seg) * DI) + c0 + lane;
    segLA[oi] = la;
    segLV[oi] = lv;
}

// --------------------------- segment combine -------------------------------
__global__ void combine_kernel(const float* __restrict__ segLA,
                               const float* __restrict__ segLV,
                               float* __restrict__ hlast) {  // [NB][DI]
    int idx = blockIdx.x * blockDim.x + threadIdx.x;
    if (idx >= NB * DI) return;
    int b = idx / DI, c = idx % DI;
    float la = 0.f, lv = -INFINITY;
    for (int s = 0; s < NSEG; ++s) {
        const float sa = segLA[(b * NSEG + s) * DI + c];
        const float sv = segLV[(b * NSEG + s) * DI + c];
        lv = logaddexp_f(lv + sa, sv);
        la += sa;
    }
    hlast[idx] = __expf(lv);
}

// --------------------------- output projection -----------------------------
__global__ void out_kernel(const float* __restrict__ hlast,  // [NB][DI]
                           const float* __restrict__ W_out,  // [DI][DD]
                           float* __restrict__ out) {        // [NB][DD]
    int idx = blockIdx.x * blockDim.x + threadIdx.x;
    if (idx >= NB * DD) return;
    int b = idx / DD, n = idx % DD;
    const float* hb = hlast + b * DI;
    float s = 0.f;
    for (int c = 0; c < DI; ++c) s += hb[c] * W_out[c * DD + n];
    out[idx] = s;
}

// ---------------------------------------------------------------------------

extern "C" void kernel_launch(void* const* d_in, const int* in_sizes, int n_in,
                              void* d_out, int out_size, void* d_ws, size_t ws_size,
                              hipStream_t stream) {
    (void)in_sizes; (void)n_in; (void)out_size; (void)ws_size;
    const float* obs   = (const float*)d_in[0];  // [16,4096,64]
    const float* W_emb = (const float*)d_in[1];  // [64,512]
    const float* b_emb = (const float*)d_in[2];  // [512]
    const float* W_hg  = (const float*)d_in[3];  // [512,1536]
    const float* W_out = (const float*)d_in[4];  // [768,512]
    float* out = (float*)d_out;                  // [16,512]

    char* ws = (char*)d_ws;
    const size_t n_obs = (size_t)NB * NS * KIN;                    // 4,194,304
    _Float16* obs16 = (_Float16*)(ws);                             // 8,388,608 B
    _Float16* Wt    = (_Float16*)(ws + 8388608);                   //   196,608 B
    float*    bcomb = (float*)   (ws + 8388608 + 196608);          //     6,144 B
    float*    segLA = (float*)   (ws + 8591360);                   //   786,432 B
    float*    segLV = (float*)   (ws + 8591360 + 786432);          //   786,432 B
    float*    hlast = (float*)   (ws + 8591360 + 2 * 786432);      //    49,152 B

    cvt_obs_kernel<<<(int)((n_obs + 255) / 256), 256, 0, stream>>>(obs, obs16, (int)n_obs);
    prep_kernel<<<2 * DI, KIN, 0, stream>>>(W_emb, b_emb, W_hg, Wt, bcomb);
    mingru_scan_kernel<<<dim3(DI / CH, NB, NSEG), 32, 0, stream>>>(obs16, Wt, bcomb, segLA, segLV);
    combine_kernel<<<(NB * DI + 255) / 256, 256, 0, stream>>>(segLA, segLV, hlast);
    out_kernel<<<(NB * DD + 255) / 256, 256, 0, stream>>>(hlast, W_out, out);
}
